// CAttention_72756745994850
// MI455X (gfx1250) — matrix operand order repaired and checked by reading, not compiled, not run on hardware
//
#include <hip/hip_runtime.h>
#include <math.h>

typedef __attribute__((ext_vector_type(16))) _Float16 v16h;
typedef __attribute__((ext_vector_type(8)))  _Float16 v8h;
typedef __attribute__((ext_vector_type(8)))  float    v8f;
typedef __attribute__((ext_vector_type(4)))  unsigned int u32x4;
typedef __attribute__((ext_vector_type(8)))  int          i32x8;
typedef __attribute__((ext_vector_type(4)))  int          i32x4;

#define WMMA_F16(A,B,C) __builtin_amdgcn_wmma_f32_16x16x32_f16(false,(A),false,(B),(short)0,(C),false,false)
#define CAT16(lo,hi) __builtin_shufflevector((lo),(hi),0,1,2,3,4,5,6,7,8,9,10,11,12,13,14,15)

// ---------------------------------------------------------------------------
// Kernel 0: convert + transpose weights to f16 (w1T: [1024][256], w2T: [256][1024])
// ---------------------------------------------------------------------------
__global__ void cvt_weights_kernel(const float* __restrict__ w1, const float* __restrict__ w2,
                                   _Float16* __restrict__ w1T, _Float16* __restrict__ w2T)
{
    int i = blockIdx.x * blockDim.x + threadIdx.x;   // 0 .. 262143
    int n1 = i >> 8, k1 = i & 255;                   // w1T[n][k] = w1[k][n], w1 is (256,1024)
    w1T[i] = (_Float16)w1[k1 * 1024 + n1];
    int n2 = i >> 10, k2 = i & 1023;                 // w2T[n][k] = w2[k][n], w2 is (1024,256)
    w2T[i] = (_Float16)w2[k2 * 256 + n2];
}

// ---------------------------------------------------------------------------
// Kernel 1: windowed attention. Block = 128 thr (4 waves) -> one (window b, head h).
// Wave rb handles queries [16rb, 16rb+16). d = 32 -> one 16x16x32 WMMA per tile.
// Writes attention output twice: f32 (for residual) and f16 (MLP GEMM operand).
// ---------------------------------------------------------------------------
__global__ __launch_bounds__(128)
void attn_kernel(const float* __restrict__ q, const float* __restrict__ k,
                 const float* __restrict__ v, const float* __restrict__ mask,
                 float* __restrict__ O32, _Float16* __restrict__ O16)
{
    __shared__ __align__(16) _Float16 Plds[4][16][64];   // 8 KB: P staging per wave
    const int b     = blockIdx.x >> 3;
    const int h     = blockIdx.x & 7;
    const int wave  = threadIdx.x >> 5;
    const int lane  = threadIdx.x & 31;
    const int half_ = lane >> 4;
    const int lq    = lane & 15;
    const int rb    = wave;

    // --- Q A-fragment (16x32 f16): lane row = lq, K = half*16 + e
    const float* qrow = q + ((size_t)(b * 64 + rb * 16 + lq)) * 256 + h * 32;
    v16h aq;
    #pragma unroll
    for (int e = 0; e < 8; ++e) {
        aq[e]     = (_Float16)qrow[half_ * 16 + e];
        aq[e + 8] = (_Float16)qrow[half_ * 16 + 8 + e];
    }

    // --- S = Q K^T : 4 column tiles (64 keys). B[c][n] = K[n][c], c = half*16 + e
    v8f s[4];
    #pragma unroll
    for (int t = 0; t < 4; ++t) {
        const float* krow = k + ((size_t)(b * 64 + t * 16 + lq)) * 256 + h * 32 + half_ * 16;
        v16h bk;
        #pragma unroll
        for (int e = 0; e < 16; ++e) bk[e] = (_Float16)krow[e];
        v8f z = {};
        s[t] = WMMA_F16(aq, bk, z);
    }

    // --- additive window mask (before scaling), then * 1/sqrt(32)
    const float* mrow = mask + (size_t)(b & 63) * 4096;
    const float scale = 0.17677669529663687f;
    #pragma unroll
    for (int t = 0; t < 4; ++t)
        #pragma unroll
        for (int r = 0; r < 8; ++r) {
            int m = rb * 16 + half_ * 8 + r;           // C-layout: row = 8*half + r
            s[t][r] = (s[t][r] + mrow[m * 64 + t * 16 + lq]) * scale;
        }

    // --- softmax over 64 keys: reduce across 4 tiles + 16 lanes of this half
    #pragma unroll
    for (int r = 0; r < 8; ++r) {
        float mx = fmaxf(fmaxf(s[0][r], s[1][r]), fmaxf(s[2][r], s[3][r]));
        #pragma unroll
        for (int off = 8; off >= 1; off >>= 1)
            mx = fmaxf(mx, __shfl_xor(mx, off, 32));
        float sum = 0.f;
        #pragma unroll
        for (int t = 0; t < 4; ++t) { float e = __expf(s[t][r] - mx); s[t][r] = e; sum += e; }
        #pragma unroll
        for (int off = 8; off >= 1; off >>= 1)
            sum += __shfl_xor(sum, off, 32);
        float inv = 1.0f / sum;
        #pragma unroll
        for (int t = 0; t < 4; ++t) s[t][r] *= inv;
    }

    // --- stage P (16x64, f16) to LDS; C-layout -> row-major
    #pragma unroll
    for (int t = 0; t < 4; ++t)
        #pragma unroll
        for (int r = 0; r < 8; ++r)
            Plds[wave][half_ * 8 + r][t * 16 + lq] = (_Float16)s[t][r];
    __syncthreads();

    // --- reload P as A-fragments (two K=32 token chunks)
    v16h ap[2];
    #pragma unroll
    for (int c = 0; c < 2; ++c) {
        const _Float16* pr = &Plds[wave][lq][c * 32];
        v8h lo = *(const v8h*)(pr + half_ * 16);
        v8h hi = *(const v8h*)(pr + half_ * 16 + 8);
        ap[c] = CAT16(lo, hi);
    }

    // --- O = P @ V : 2 output tiles (d=32), K chained over two token chunks
    v8f o[2] = {};
    #pragma unroll
    for (int c = 0; c < 2; ++c) {
        #pragma unroll
        for (int u = 0; u < 2; ++u) {
            const float* vp = v + ((size_t)(b * 64 + c * 32 + half_ * 16)) * 256 + h * 32 + u * 16 + lq;
            v16h bv;
            #pragma unroll
            for (int e = 0; e < 16; ++e) bv[e] = (_Float16)vp[(size_t)e * 256];
            o[u] = WMMA_F16(ap[c], bv, o[u]);
        }
    }

    // --- store attention output (b, token, h*32 + dcol): f32 + f16 copies
    #pragma unroll
    for (int u = 0; u < 2; ++u)
        #pragma unroll
        for (int r = 0; r < 8; ++r) {
            size_t idx = ((size_t)(b * 64 + rb * 16 + half_ * 8 + r)) * 256 + h * 32 + u * 16 + lq;
            O32[idx] = o[u][r];
            O16[idx] = (_Float16)o[u][r];
        }
}

// ---------------------------------------------------------------------------
// Kernel 2: fused MLP. Block = 256 thr (8 waves) -> 64 token rows.
// X tile DMA'd into LDS by the Tensor Data Mover (with LDS row padding),
// GEMM1 (K=256) -> bias + exact GELU -> H tile (64x1024 f16) in LDS ->
// GEMM2 (K=1024) -> + b2 + residual.
// ---------------------------------------------------------------------------
__global__ __launch_bounds__(256)
void mlp_kernel(const float* __restrict__ X32, const _Float16* __restrict__ X16,
                const _Float16* __restrict__ w1T, const float* __restrict__ b1,
                const _Float16* __restrict__ w2T, const float* __restrict__ b2,
                float* __restrict__ out)
{
    __shared__ __align__(16) _Float16 Xlds[64 * 264];  // 33 KB: 64 rows, pitch 528B (512B data + 16B pad)
    __shared__ __align__(16) _Float16 Hlds[64][1032];  // 132 KB, +8 halves row pad
    const int R0    = blockIdx.x * 64;
    const int wave  = threadIdx.x >> 5;
    const int lane  = threadIdx.x & 31;
    const int half_ = lane >> 4;
    const int lq    = lane & 15;

    // ---- TDM: DMA 64x256 f16 tile of X into LDS (one descriptor, wave 0 issues) ----
    if (threadIdx.x < 32) {
        unsigned lds0 = (unsigned)(uintptr_t)&Xlds[0];                    // LDS byte offset
        unsigned long long ga = (unsigned long long)(uintptr_t)(X16 + (size_t)R0 * 256);
        // D# group0: count=1 | lds_addr | global_addr[56:0] | type=2
        u32x4 g0 = { 1u, lds0, (unsigned)ga,
                     ((unsigned)(ga >> 32) & 0x01FFFFFFu) | 0x80000000u };
        // D# group1: data_size=2B, pad_enable, pad_interval=128dw(512B), pad_amount=4dw(16B)
        //            tensor_dim0=256, tensor_dim1=65536, tile_dim0=256, tile_dim1=64,
        //            tensor_dim0_stride=256
        i32x8 g1 = { (int)0x07910000u,      // wgmask=0,dsize=1,pad_en,pi=6,pa=3
                     (int)(256u << 16),     // atomic_addr=0 | tensor_dim0.lo=256
                     0,                     // tensor_dim0.hi=0 | tensor_dim1.lo=0
                     (int)((256u << 16) | 1u), // tensor_dim1.hi=1 | tile_dim0=256
                     64,                    // tile_dim1=64 | tile_dim2=0
                     256,                   // tensor_dim0_stride=256
                     0, 0 };
        // benign dims for groups 2/3 (tensor_dim2=1, tensor_dim3=1, tensor_dim4=1)
        i32x4 g2 = { 1, 1, 0, 0 };
        i32x4 g3 = { 0, (int)(1u << 16), 0, 0 };
#if __clang_major__ >= 23
        i32x8 gz8 = {};
        __builtin_amdgcn_tensor_load_to_lds(g0, g1, g2, g3, gz8, 0);
#else
        __builtin_amdgcn_tensor_load_to_lds(g0, g1, g2, g3, 0);
#endif
        __builtin_amdgcn_s_wait_tensorcnt(0);
    }
    __syncthreads();

    // ---- GEMM1: wave owns 8 hidden col-tiles, in 2 passes of 4 (reg pressure) ----
    for (int p = 0; p < 2; ++p) {
        v8f acc[4][4] = {};
        for (int kc = 0; kc < 8; ++kc) {
            v16h ax[4];
            #pragma unroll
            for (int rt = 0; rt < 4; ++rt) {
                const _Float16* xr = Xlds + (rt * 16 + lq) * 264 + kc * 32;
                v8h lo = *(const v8h*)(xr + half_ * 16);
                v8h hi = *(const v8h*)(xr + half_ * 16 + 8);
                ax[rt] = CAT16(lo, hi);
            }
            #pragma unroll
            for (int cc = 0; cc < 4; ++cc) {
                const int ct = wave * 8 + p * 4 + cc;
                v16h bw = *(const v16h*)(w1T + (size_t)(ct * 16 + lq) * 256 + kc * 32 + half_ * 16);
                #pragma unroll
                for (int rt = 0; rt < 4; ++rt)
                    acc[rt][cc] = WMMA_F16(ax[rt], bw, acc[rt][cc]);
            }
        }
        // epilogue: bias + exact GELU -> H (f16) in LDS
        #pragma unroll
        for (int cc = 0; cc < 4; ++cc) {
            const int ct = wave * 8 + p * 4 + cc;
            const float bias = b1[ct * 16 + lq];
            #pragma unroll
            for (int rt = 0; rt < 4; ++rt)
                #pragma unroll
                for (int r = 0; r < 8; ++r) {
                    float x = acc[rt][cc][r] + bias;
                    float g = 0.5f * x * (1.0f + erff(x * 0.70710678118654752f));
                    Hlds[rt * 16 + half_ * 8 + r][ct * 16 + lq] = (_Float16)g;
                }
        }
    }
    __syncthreads();

    // ---- GEMM2: wave owns 2 output col-tiles x 4 row-tiles; K = 1024 ----
    v8f acc2[2][4] = {};
    for (int kc = 0; kc < 32; ++kc) {
        v16h ah[4];
        #pragma unroll
        for (int rt = 0; rt < 4; ++rt) {
            const _Float16* hr = &Hlds[rt * 16 + lq][kc * 32];
            v8h lo = *(const v8h*)(hr + half_ * 16);
            v8h hi = *(const v8h*)(hr + half_ * 16 + 8);
            ah[rt] = CAT16(lo, hi);
        }
        #pragma unroll
        for (int c2 = 0; c2 < 2; ++c2) {
            const int ct = wave * 2 + c2;
            v16h bw = *(const v16h*)(w2T + (size_t)(ct * 16 + lq) * 1024 + kc * 32 + half_ * 16);
            #pragma unroll
            for (int rt = 0; rt < 4; ++rt)
                acc2[c2][rt] = WMMA_F16(ah[rt], bw, acc2[c2][rt]);
        }
    }
    // epilogue: + b2 + residual X (f32 copy for precision)
    #pragma unroll
    for (int c2 = 0; c2 < 2; ++c2) {
        const int ct = wave * 2 + c2;
        const float bias = b2[ct * 16 + lq];
        #pragma unroll
        for (int rt = 0; rt < 4; ++rt)
            #pragma unroll
            for (int r = 0; r < 8; ++r) {
                size_t idx = ((size_t)(R0 + rt * 16 + half_ * 8 + r)) * 256 + ct * 16 + lq;
                out[idx] = X32[idx] + acc2[c2][rt][r] + bias;
            }
    }
}

// ---------------------------------------------------------------------------
extern "C" void kernel_launch(void* const* d_in, const int* in_sizes, int n_in,
                              void* d_out, int out_size, void* d_ws, size_t ws_size,
                              hipStream_t stream) {
    const float* q    = (const float*)d_in[0];
    const float* k    = (const float*)d_in[1];
    const float* v    = (const float*)d_in[2];
    const float* mask = (const float*)d_in[3];
    // d_in[4], d_in[5] are H, W scalars (unused)
    const float* w1   = (const float*)d_in[6];
    const float* b1   = (const float*)d_in[7];
    const float* w2   = (const float*)d_in[8];
    const float* b2   = (const float*)d_in[9];
    float* out = (float*)d_out;

    char* ws = (char*)d_ws;
    float*    O32 = (float*)ws;                              // 64 MB
    _Float16* O16 = (_Float16*)(ws + 67108864);              // 32 MB
    _Float16* w1T = (_Float16*)(ws + 67108864 + 33554432);   // 512 KB
    _Float16* w2T = (_Float16*)(ws + 67108864 + 33554432 + 524288);

    cvt_weights_kernel<<<1024, 256, 0, stream>>>(w1, w2, w1T, w2T);
    attn_kernel<<<8192, 128, 0, stream>>>(q, k, v, mask, O32, O16);
    mlp_kernel<<<1024, 256, 0, stream>>>(O32, O16, w1T, b1, w2T, b2, out);
}
